// CrossRegionRelationalReasoning_53309134078400
// MI455X (gfx1250) — matrix operand order repaired and checked by reading, not compiled, not run on hardware
//
#include <hip/hip_runtime.h>
#include <hip/hip_bf16.h>
#include <math.h>

// Problem constants (from reference)
#define BATCH  4096
#define NREG   5
#define CIN    512
#define HIDN   256
#define HEADS  4
#define HD     64          // HIDN / HEADS
#define LAYERS 3
#define ROWS   (BATCH * NREG)   // 20480

typedef _Float16 h8  __attribute__((ext_vector_type(8)));
typedef _Float16 h16 __attribute__((ext_vector_type(16)));
typedef float    f8  __attribute__((ext_vector_type(8)));

// ---------------------------------------------------------------------------
// 1) adaptive_avg_pool2d(1) : [B,N,C,4,4] f32 -> node [B*N, C] f16
//    Each thread reduces its 16 contiguous floats (64B contiguous per lane).
// ---------------------------------------------------------------------------
__global__ void __launch_bounds__(256)
pool_kernel(const float* __restrict__ rf, _Float16* __restrict__ node16) {
    int idx = blockIdx.x * blockDim.x + threadIdx.x;   // [0, ROWS*CIN)
    if (idx >= ROWS * CIN) return;
    const float4* p = (const float4*)(rf + (size_t)idx * 16);
    float4 a = p[0], b = p[1], c = p[2], d = p[3];
    float s = (a.x + a.y + a.z + a.w) + (b.x + b.y + b.z + b.w) +
              (c.x + c.y + c.z + c.w) + (d.x + d.y + d.z + d.w);
    node16[idx] = (_Float16)(s * (1.0f / 16.0f));
}

// ---------------------------------------------------------------------------
// 2) generic f32 -> f16 convert
// ---------------------------------------------------------------------------
__global__ void __launch_bounds__(256)
f32_to_f16_kernel(const float* __restrict__ src, _Float16* __restrict__ dst, int n) {
    int i = blockIdx.x * blockDim.x + threadIdx.x;
    if (i < n) dst[i] = (_Float16)src[i];
}

// ---------------------------------------------------------------------------
// 3) WMMA GEMM: out[M,N] = A16[M,K] @ B16[K,N] + bias[N]   (f32 accumulate)
//    Register-blocked: one wave computes a 32x64 output block
//    (2 M-tiles x 4 N-tiles, 8 f32 accumulators). Per 32-K step:
//    4 A-loads (b128) + 64 strided 16-bit B-loads feed 8 v_wmma_f32_16x16x32_f16.
//    Fragment layouts per CDNA5 ISA 7.12.2:
//      A (16x32 f16): lanes 0-15 row M=lane, halves -> K {0..7,16..23};
//                     lanes 16-31 same rows, K {8..15,24..31}.
//      B (32x16 f16): lane L holds col N=L%16, halves -> same K as A lane L.
//      C/D (16x16 f32): lane<16 -> N=lane, VGPR v = row M=v;
//                       lane>=16 -> N=lane-16, rows M=8..15.
// ---------------------------------------------------------------------------
__global__ void __launch_bounds__(128)
wmma_gemm_f16(const _Float16* __restrict__ A, const _Float16* __restrict__ Bw,
              const float* __restrict__ bias, float* __restrict__ out,
              int M, int N, int K) {
    int lane = threadIdx.x & 31;
    int wave = threadIdx.x >> 5;
    int tile = blockIdx.x * 4 + wave;
    int tilesN = N >> 6;                   // 64-wide column blocks
    int tm = tile / tilesN;
    int tn = tile - tm * tilesN;
    if (tm * 32 >= M) return;              // wave-uniform (EXEC stays all-1)
    int m0 = tm * 32, n0 = tn * 64;
    int laneLo = lane & 15;
    int hiOff  = (lane >= 16) ? 8 : 0;

    const _Float16* arow0 = A + (size_t)(m0 + laneLo) * K + hiOff;   // rows m0..m0+15
    const _Float16* arow1 = arow0 + (size_t)16 * K;                  // rows m0+16..m0+31
    const _Float16* brow  = Bw + (size_t)hiOff * N + n0 + laneLo;    // col n0+laneLo, row k0+hiOff

    f8 acc[8] = {};
    for (int k0 = 0; k0 < K; k0 += 32) {
        union { h16 v; h8 h[2]; } a0, a1;
        a0.h[0] = *(const h8*)(arow0 + k0);          // K = k0+hiOff+0..7
        a0.h[1] = *(const h8*)(arow0 + k0 + 16);     // K = k0+16+hiOff+0..7
        a1.h[0] = *(const h8*)(arow1 + k0);
        a1.h[1] = *(const h8*)(arow1 + k0 + 16);
        const _Float16* bk = brow + (size_t)k0 * N;
#pragma unroll
        for (int c = 0; c < 4; ++c) {
            union { h16 v; h8 h[2]; } bf;
            for (int s = 0; s < 8; ++s) {
                bf.h[0][s] = bk[(size_t)s * N + c * 16];          // K = k0+hiOff+s
                bf.h[1][s] = bk[(size_t)(s + 16) * N + c * 16];   // K = k0+16+hiOff+s
            }
            acc[2 * c] = __builtin_amdgcn_wmma_f32_16x16x32_f16(
                false, a0.v, false, bf.v, (short)0, acc[2 * c], false, false);
            acc[2 * c + 1] = __builtin_amdgcn_wmma_f32_16x16x32_f16(
                false, a1.v, false, bf.v, (short)0, acc[2 * c + 1], false, false);
        }
    }

    int rbase = m0 + ((lane >= 16) ? 8 : 0);
#pragma unroll
    for (int c = 0; c < 4; ++c) {
        float bcol = bias[n0 + c * 16 + laneLo];
        float* op = out + (size_t)rbase * N + n0 + c * 16 + laneLo;
#pragma unroll
        for (int v = 0; v < 8; ++v) {
            op[(size_t)v * N]        = acc[2 * c][v] + bcol;       // rows rbase+v
            op[(size_t)(v + 16) * N] = acc[2 * c + 1][v] + bcol;   // rows rbase+16+v
        }
    }
}

// ---------------------------------------------------------------------------
// 4) attention: per (batch, head) one wave. Each lane owns head-dims
//    {lane, lane+32}. 5x5 scores via shuffle butterfly reduction, softmax
//    (with adjacency mask), context, attn probs -> d_out slice.
// ---------------------------------------------------------------------------
__global__ void __launch_bounds__(128)
attn_kernel(const float* __restrict__ Q, const float* __restrict__ Kb,
            const float* __restrict__ V, const float* __restrict__ adj,
            float* __restrict__ ctx, float* __restrict__ attn_out) {
    int b    = blockIdx.x;
    int h    = threadIdx.x >> 5;
    int lane = threadIdx.x & 31;
    int col0 = h * HD;

    float qx[NREG], qy[NREG], kx[NREG], ky[NREG], vx[NREG], vy[NREG];
#pragma unroll
    for (int n = 0; n < NREG; ++n) {
        size_t off = ((size_t)(b * NREG + n)) * HIDN + col0 + lane;
        qx[n] = Q[off];  qy[n] = Q[off + 32];
        kx[n] = Kb[off]; ky[n] = Kb[off + 32];
        vx[n] = V[off];  vy[n] = V[off + 32];
    }

    const float scale = 0.125f;   // 1/sqrt(HD=64)
    float s[NREG][NREG];
#pragma unroll
    for (int n = 0; n < NREG; ++n) {
#pragma unroll
        for (int m = 0; m < NREG; ++m) {
            float p = qx[n] * kx[m] + qy[n] * ky[m];
#pragma unroll
            for (int o = 16; o >= 1; o >>= 1) p += __shfl_xor(p, o, 32);
            p *= scale;
            if (adj[n * NREG + m] == 0.0f) p = -__builtin_inff();
            s[n][m] = p;
        }
    }

    float a[NREG][NREG];
#pragma unroll
    for (int n = 0; n < NREG; ++n) {
        float mx = s[n][0];
#pragma unroll
        for (int m = 1; m < NREG; ++m) mx = fmaxf(mx, s[n][m]);
        float sum = 0.0f;
#pragma unroll
        for (int m = 0; m < NREG; ++m) { a[n][m] = __expf(s[n][m] - mx); sum += a[n][m]; }
        float inv = 1.0f / sum;
#pragma unroll
        for (int m = 0; m < NREG; ++m) a[n][m] *= inv;
    }

    if (lane == 0) {
        float* ap = attn_out + (((size_t)b * HEADS + h) * NREG) * NREG;
#pragma unroll
        for (int n = 0; n < NREG; ++n)
#pragma unroll
            for (int m = 0; m < NREG; ++m) ap[n * NREG + m] = a[n][m];
    }

#pragma unroll
    for (int n = 0; n < NREG; ++n) {
        float cx = 0.0f, cy = 0.0f;
#pragma unroll
        for (int m = 0; m < NREG; ++m) { cx += a[n][m] * vx[m]; cy += a[n][m] * vy[m]; }
        size_t off = ((size_t)(b * NREG + n)) * HIDN + col0 + lane;
        ctx[off]      = cx;
        ctx[off + 32] = cy;
    }
}

// ---------------------------------------------------------------------------
// 5) residual add + LayerNorm (biased var, eps=1e-5). One wave per row.
// ---------------------------------------------------------------------------
__global__ void __launch_bounds__(128)
ln_residual_kernel(const float* __restrict__ proj, const float* __restrict__ res,
                   const float* __restrict__ g, const float* __restrict__ bb,
                   float* __restrict__ xout) {
    int wave = threadIdx.x >> 5;
    int lane = threadIdx.x & 31;
    int row  = blockIdx.x * 4 + wave;
    if (row >= ROWS) return;
    const float* pr = proj + (size_t)row * HIDN;
    const float* rr = res  + (size_t)row * HIDN;
    float y[8];
    float s1 = 0.0f, s2 = 0.0f;
#pragma unroll
    for (int t = 0; t < 8; ++t) {
        int c = lane + 32 * t;
        y[t] = pr[c] + rr[c];
        s1 += y[t];
        s2 += y[t] * y[t];
    }
#pragma unroll
    for (int o = 16; o >= 1; o >>= 1) {
        s1 += __shfl_xor(s1, o, 32);
        s2 += __shfl_xor(s2, o, 32);
    }
    float mu   = s1 * (1.0f / HIDN);
    float var  = s2 * (1.0f / HIDN) - mu * mu;
    float rstd = rsqrtf(var + 1e-5f);
    float* xo = xout + (size_t)row * HIDN;
#pragma unroll
    for (int t = 0; t < 8; ++t) {
        int c = lane + 32 * t;
        xo[c] = (y[t] - mu) * rstd * g[c] + bb[c];
    }
}

// ---------------------------------------------------------------------------
// launch
// ---------------------------------------------------------------------------
extern "C" void kernel_launch(void* const* d_in, const int* in_sizes, int n_in,
                              void* d_out, int out_size, void* d_ws, size_t ws_size,
                              hipStream_t stream) {
    (void)in_sizes; (void)n_in; (void)out_size; (void)ws_size;
    const float* rf   = (const float*)d_in[0];
    const float* adj  = (const float*)d_in[1];
    const float* w_in = (const float*)d_in[2];
    const float* b_in = (const float*)d_in[3];
    const float* wq   = (const float*)d_in[4];
    const float* bq   = (const float*)d_in[5];
    const float* wk   = (const float*)d_in[6];
    const float* bk   = (const float*)d_in[7];
    const float* wv   = (const float*)d_in[8];
    const float* bv   = (const float*)d_in[9];
    const float* wo   = (const float*)d_in[10];
    const float* bo   = (const float*)d_in[11];
    const float* lng  = (const float*)d_in[12];
    const float* lnb  = (const float*)d_in[13];

    float* out_x    = (float*)d_out;                          // [ROWS, HIDN]
    float* out_attn = out_x + (size_t)ROWS * HIDN;            // [B, HEADS, 5, 5]

    // bump allocator over workspace (~170 MB total)
    char* ws = (char*)d_ws;
    size_t off = 0;
    auto alloc = [&](size_t bytes) -> void* {
        void* p = ws + off;
        off = (off + bytes + 255) & ~(size_t)255;
        return p;
    };
    _Float16* node16 = (_Float16*)alloc((size_t)ROWS * CIN * 2);
    _Float16* w_in16 = (_Float16*)alloc((size_t)CIN * HIDN * 2);
    _Float16* wq16   = (_Float16*)alloc((size_t)LAYERS * HIDN * HIDN * 2);
    _Float16* wk16   = (_Float16*)alloc((size_t)LAYERS * HIDN * HIDN * 2);
    _Float16* wv16   = (_Float16*)alloc((size_t)LAYERS * HIDN * HIDN * 2);
    _Float16* wo16   = (_Float16*)alloc((size_t)LAYERS * HIDN * HIDN * 2);
    float*    x      = (float*)   alloc((size_t)ROWS * HIDN * 4);
    _Float16* x16    = (_Float16*)alloc((size_t)ROWS * HIDN * 2);
    float*    qb     = (float*)   alloc((size_t)ROWS * HIDN * 4);
    float*    kb     = (float*)   alloc((size_t)ROWS * HIDN * 4);
    float*    vb     = (float*)   alloc((size_t)ROWS * HIDN * 4);
    float*    ctx    = (float*)   alloc((size_t)ROWS * HIDN * 4);
    _Float16* ctx16  = (_Float16*)alloc((size_t)ROWS * HIDN * 2);
    float*    ob     = (float*)   alloc((size_t)ROWS * HIDN * 4);

    // 1) pooling (HBM-bound: 671 MB streamed)
    pool_kernel<<<(ROWS * CIN) / 256, 256, 0, stream>>>(rf, node16);

    // 2) one-time weight conversion to f16
    f32_to_f16_kernel<<<(CIN * HIDN + 255) / 256, 256, 0, stream>>>(w_in, w_in16, CIN * HIDN);
    const int wsz = LAYERS * HIDN * HIDN;
    f32_to_f16_kernel<<<(wsz + 255) / 256, 256, 0, stream>>>(wq, wq16, wsz);
    f32_to_f16_kernel<<<(wsz + 255) / 256, 256, 0, stream>>>(wk, wk16, wsz);
    f32_to_f16_kernel<<<(wsz + 255) / 256, 256, 0, stream>>>(wv, wv16, wsz);
    f32_to_f16_kernel<<<(wsz + 255) / 256, 256, 0, stream>>>(wo, wo16, wsz);

    const int tiles  = (ROWS / 32) * (HIDN / 64);   // 2560 wave-blocks (32x64 each)
    const int gblks  = tiles / 4;                   // 4 waves / block -> 640 blocks

    // 3) input projection: x = node @ w_in + b_in  (K=512)
    wmma_gemm_f16<<<gblks, 128, 0, stream>>>(node16, w_in16, b_in, x, ROWS, HIDN, CIN);

    // 4) transformer layers
    for (int i = 0; i < LAYERS; ++i) {
        f32_to_f16_kernel<<<(ROWS * HIDN + 255) / 256, 256, 0, stream>>>(x, x16, ROWS * HIDN);
        wmma_gemm_f16<<<gblks, 128, 0, stream>>>(x16, wq16 + (size_t)i * HIDN * HIDN,
                                                 bq + i * HIDN, qb, ROWS, HIDN, HIDN);
        wmma_gemm_f16<<<gblks, 128, 0, stream>>>(x16, wk16 + (size_t)i * HIDN * HIDN,
                                                 bk + i * HIDN, kb, ROWS, HIDN, HIDN);
        wmma_gemm_f16<<<gblks, 128, 0, stream>>>(x16, wv16 + (size_t)i * HIDN * HIDN,
                                                 bv + i * HIDN, vb, ROWS, HIDN, HIDN);
        attn_kernel<<<BATCH, 128, 0, stream>>>(qb, kb, vb, adj, ctx, out_attn);
        f32_to_f16_kernel<<<(ROWS * HIDN + 255) / 256, 256, 0, stream>>>(ctx, ctx16, ROWS * HIDN);
        wmma_gemm_f16<<<gblks, 128, 0, stream>>>(ctx16, wo16 + (size_t)i * HIDN * HIDN,
                                                 bo + i * HIDN, ob, ROWS, HIDN, HIDN);
        ln_residual_kernel<<<ROWS / 4, 128, 0, stream>>>(
            ob, x, lng + i * HIDN, lnb + i * HIDN,
            (i == LAYERS - 1) ? out_x : x);
    }
}
